// LSTM_discriminator_10204842295651
// MI455X (gfx1250) — matrix operand order repaired and checked by reading, not compiled, not run on hardware
//
#include <hip/hip_runtime.h>
#include <hip/hip_bf16.h>

typedef __attribute__((ext_vector_type(16))) __bf16 v16bf;
typedef __attribute__((ext_vector_type(8)))  __bf16 v8bf;
typedef __attribute__((ext_vector_type(8)))  float  v8f;

#define B_TOT 2048
#define T_LEN 512
#define H_DIM 128
#define BM    16      // batch rows per workgroup
#define NTHREADS 256  // 8 wave32s

// ---- branch-free fast activations (gfx1250 has native V_TANH_F32) ----
__device__ __forceinline__ float fast_tanh(float x) {
#if __has_builtin(__builtin_amdgcn_tanhf)
    return __builtin_amdgcn_tanhf(x);
#else
    float e = __builtin_amdgcn_exp2f(x * 2.885390081777927f); // 2*log2(e)
    return 1.0f - 2.0f * __builtin_amdgcn_rcpf(e + 1.0f);
#endif
}

// sigmoid(x) given y = 0.5*x (the 0.5 is pre-folded into weights/bias)
__device__ __forceinline__ float sigmoid_from_half(float y) {
    return __builtin_fmaf(fast_tanh(y), 0.5f, 0.5f);
}

__device__ __forceinline__ void sched_fence() {
#if __has_builtin(__builtin_amdgcn_sched_barrier)
    __builtin_amdgcn_sched_barrier(0);
#endif
}

__global__ __launch_bounds__(NTHREADS, 1)
void lstm_disc_kernel(const float* __restrict__ x,     // [B,T]
                      const float* __restrict__ hx0,   // [B,H]
                      const float* __restrict__ cx0,   // [B,H]
                      const float* __restrict__ W_ih,  // [4H,1]
                      const float* __restrict__ W_hh,  // [4H,H]
                      const float* __restrict__ b_ih,  // [4H]
                      const float* __restrict__ b_hh,  // [4H]
                      const float* __restrict__ W_mlp, // [1,H]
                      const float* __restrict__ b_mlp, // [1]
                      float* __restrict__ out)         // [B,1]
{
    __shared__ float  xT[T_LEN][BM];           // 32 KB, x transposed for this batch tile
    __shared__ __bf16 hxL[2][BM][H_DIM];       // 8 KB ping-pong recurrent state (bf16)

    const int tid  = threadIdx.x;
    const int w    = tid >> 5;                 // wave id 0..7 -> hidden slice
    const int lane = tid & 31;
    const int half = lane >> 4;                // lane group 0/1
    const int ln   = lane & 15;
    const int b0   = blockIdx.x * BM;
    const int hcol = 16 * w + ln;              // this lane's hidden column

    // ---- stage x transposed into LDS (coalesced: consecutive tid -> consecutive t) ----
    for (int i = tid; i < BM * T_LEN; i += NTHREADS) {
        int m = i / T_LEN, t = i % T_LEN;
        xT[t][m] = x[(size_t)(b0 + m) * T_LEN + t];
    }

    // ---- init hx (LDS, bf16) and cx (registers, C/D fragment layout) ----
    v8f cxv;
    #pragma unroll
    for (int r = 0; r < 8; ++r) {
        int m = r + 8 * half;
        hxL[0][m][hcol] = (__bf16)hx0[(size_t)(b0 + m) * H_DIM + hcol];
        cxv[r] = cx0[(size_t)(b0 + m) * H_DIM + hcol];
    }

    // ---- per-lane gate input weight / bias (input dim == 1) ----
    // For sigmoid gates (i=0, f=1, o=3) fold the 0.5 of sigmoid(x)=0.5*tanh(x/2)+0.5
    // into the weights/bias so the hot loop needs no extra multiplies.
    float wn[4], bs[4];
    #pragma unroll
    for (int g = 0; g < 4; ++g) {
        const float scale = (g == 2) ? 1.0f : 0.5f;
        int n = g * H_DIM + hcol;
        wn[g] = W_ih[n] * scale;
        bs[g] = (b_ih[n] + b_hh[n]) * scale;
    }

    // ---- W_hh -> register-resident bf16 B-fragments, contiguous k order (same as A):
    //      fragment slot j holds K = kb*32 + half*16 + j, N = hcol.
    //      Sigmoid-gate rows pre-scaled by 0.5 (see above). ----
    v16bf bfrag[4][4];
    #pragma unroll
    for (int g = 0; g < 4; ++g) {
        const float scale = (g == 2) ? 1.0f : 0.5f;
        const float* wrow = W_hh + (size_t)(g * H_DIM + hcol) * H_DIM;
        #pragma unroll
        for (int kb = 0; kb < 4; ++kb) {
            const int kbase = kb * 32 + half * 16;
            v16bf f;
            #pragma unroll
            for (int j = 0; j < 16; ++j) f[j] = (__bf16)(wrow[kbase + j] * scale);
            bfrag[g][kb] = f;
        }
    }
    __syncthreads();

    // ================= recurrent scan =================
    for (int t = 0; t < T_LEN; ++t) {
        __bf16 (*hr)[H_DIM] = hxL[t & 1];

        // A-fragments from LDS (layout: slot j -> K=16*half+j, matching B fragments).
        // Issue all 8 ds_load_b128 before any WMMA (sched fence keeps them hoisted).
        v16bf afr[4];
        #pragma unroll
        for (int kb = 0; kb < 4; ++kb) {
            const __bf16* p = &hr[ln][kb * 32 + half * 16];
            v8bf lo = *(const v8bf*)(p);        // ds_load_b128
            v8bf hi = *(const v8bf*)(p + 8);    // ds_load_b128
            v16bf a;
            #pragma unroll
            for (int j = 0; j < 8; ++j) { a[j] = lo[j]; a[j + 8] = hi[j]; }
            afr[kb] = a;
        }
        sched_fence();

        // input contribution seeds the accumulator: gx = x[b,t]*w + bias (pre-scaled)
        const float* xp = &xT[t][half * 8];
        v8f acc[4];
        #pragma unroll
        for (int g = 0; g < 4; ++g) {
            v8f c;
            #pragma unroll
            for (int r = 0; r < 8; ++r) c[r] = __builtin_fmaf(xp[r], wn[g], bs[g]);
            #pragma unroll
            for (int kb = 0; kb < 4; ++kb)
                c = __builtin_amdgcn_wmma_f32_16x16x32_bf16(
                        /*neg_a=*/false, afr[kb],
                        /*neg_b=*/false, bfrag[g][kb],
                        /*c_mod=*/(short)0, c,
                        /*reuse_a=*/false, /*reuse_b=*/false);
            acc[g] = c;
        }

        // gate nonlinearities + state update (wave-local, branch-free trans ops)
        __bf16 (*hw)[H_DIM] = hxL[(t + 1) & 1];
        #pragma unroll
        for (int r = 0; r < 8; ++r) {
            float ig = sigmoid_from_half(acc[0][r]);   // weights pre-scaled by 0.5
            float fg = sigmoid_from_half(acc[1][r]);
            float gg = fast_tanh(acc[2][r]);           // unscaled
            float og = sigmoid_from_half(acc[3][r]);
            float cn = __builtin_fmaf(fg, cxv[r], ig * gg);
            cxv[r] = cn;
            hw[r + 8 * half][hcol] = (__bf16)(og * fast_tanh(cn));
        }
        __syncthreads();   // publish hx for next step (ping-pong, one barrier/step)
    }

    // ---- epilogue: out = sigmoid(hx @ W_mlp.T + b_mlp)  (T even -> final hx in buffer 0) ----
    if (tid < BM) {
        const __bf16* hv = hxL[0][tid];
        float s = b_mlp[0];
        #pragma unroll 8
        for (int h = 0; h < H_DIM; ++h) s = __builtin_fmaf((float)hv[h], W_mlp[h], s);
        out[b0 + tid] = sigmoid_from_half(0.5f * s);
    }
}

extern "C" void kernel_launch(void* const* d_in, const int* in_sizes, int n_in,
                              void* d_out, int out_size, void* d_ws, size_t ws_size,
                              hipStream_t stream) {
    const float* x     = (const float*)d_in[0];
    const float* hx0   = (const float*)d_in[1];
    const float* cx0   = (const float*)d_in[2];
    const float* W_ih  = (const float*)d_in[3];
    const float* W_hh  = (const float*)d_in[4];
    const float* b_ih  = (const float*)d_in[5];
    const float* b_hh  = (const float*)d_in[6];
    const float* W_mlp = (const float*)d_in[7];
    const float* b_mlp = (const float*)d_in[8];
    float* out = (float*)d_out;

    dim3 grid(B_TOT / BM);
    dim3 block(NTHREADS);
    lstm_disc_kernel<<<grid, block, 0, stream>>>(x, hx0, cx0, W_ih, W_hh,
                                                 b_ih, b_hh, W_mlp, b_mlp, out);
}
